// CrossAttention_24541443129647
// MI455X (gfx1250) — matrix operand order repaired and checked by reading, not compiled, not run on hardware
//
#include <hip/hip_runtime.h>
#include <hip/hip_bf16.h>

typedef __attribute__((ext_vector_type(16))) _Float16 v16h;
typedef __attribute__((ext_vector_type(8)))  _Float16 v8h;
typedef __attribute__((ext_vector_type(8)))  float    v8f;

#define NQ 8192
#define NK 8192
#define DIM 256
#define NSPLIT 8                                // K-column splits for occupancy
#define TILES_PER_SPLIT ((NK / 16) / NSPLIT)    // 64
#define LDSW 264                                // padded row stride (halfs): bank-conflict-free

// ---------------------------------------------------------------------------
// Kernel 0: transpose + convert Wq/Wk (f32 row-major) -> Wt[n][k] (f16).
// Makes every later B-fragment a single contiguous 32B load.
// ---------------------------------------------------------------------------
__global__ __launch_bounds__(256) void wconv_kernel(
    const float* __restrict__ Wq, const float* __restrict__ Wk,
    _Float16* __restrict__ Wqt, _Float16* __restrict__ Wkt)
{
    const int k = blockIdx.x & 255;                       // source row
    const float*    W  = (blockIdx.x < 256) ? Wq  : Wk;
    _Float16*       Wt = (blockIdx.x < 256) ? Wqt : Wkt;
    const int n = threadIdx.x;
    Wt[(size_t)n * DIM + k] = (_Float16)W[(size_t)k * DIM + n];   // coalesced read
}

// ---------------------------------------------------------------------------
// Kernel 1: projections Q = xq@Wq, K = xk@Wk (f32 in, f16 out, f32 accum).
// One wave computes a 16x16 output tile via 8x V_WMMA_F32_16X16X32_F16.
// ---------------------------------------------------------------------------
__global__ __launch_bounds__(256) void proj_kernel(
    const float* __restrict__ xq, const float* __restrict__ xk,
    const _Float16* __restrict__ Wqt, const _Float16* __restrict__ Wkt,
    _Float16* __restrict__ Qh, _Float16* __restrict__ Kh)
{
    const int lane = threadIdx.x & 31;
    const int wv   = threadIdx.x >> 5;
    const int lg   = lane >> 4;        // 16-lane group (0/1)
    const int ln   = lane & 15;

    int wtile   = blockIdx.x * 8 + wv;     // 0..16383
    int colTile = wtile & 15;              // 256/16 col tiles
    int row     = (wtile >> 4) * 16;       // over concat(Q rows, K rows)

    const float* X; const _Float16* Wt; _Float16* Y;
    if (row < NQ) { X = xq; Wt = Wqt; Y = Qh; }
    else          { X = xk; Wt = Wkt; Y = Kh; row -= NQ; }
    const int colbase = colTile * 16;

    v8f c = {};
#pragma unroll
    for (int f = 0; f < 8; ++f) {
        // A fragment: X rows [row..row+16), k = f*32 + lg*16 + e (contiguous)
        v16h a;
        const float* arow = X + (size_t)(row + ln) * DIM + f * 32 + lg * 16;
#pragma unroll
        for (int e = 0; e < 8; ++e) a[e]     = (_Float16)arow[e];
#pragma unroll
        for (int e = 0; e < 8; ++e) a[8 + e] = (_Float16)arow[8 + e];
        // B fragment: Wt[n][k], n = colbase+ln, k = f*32 + lg*16 + e (contiguous)
        v16h b = *(const v16h*)(Wt + (size_t)(colbase + ln) * DIM + f * 32 + lg * 16);

        c = __builtin_amdgcn_wmma_f32_16x16x32_f16(
                false, a, false, b, (short)0, c, false, false);
    }
    // D layout: element r -> row (row + r + 8*lg), col (colbase + ln)
#pragma unroll
    for (int r = 0; r < 8; ++r)
        Y[(size_t)(row + r + lg * 8) * DIM + colbase + ln] = (_Float16)c[r];
}

// ---------------------------------------------------------------------------
// Kernel 2: fused scores + online logsumexp (flash style), software-pipelined.
// Block = 8 waves; each wave owns 16 Q rows (A fragments VGPR-resident).
// K tiles: loads for tile nt+1 are issued before computing tile nt, so the
// 8 WMMAs overlap the L2 round-trip; LDS publish waits only at the next
// iteration's ds_store. Hot loop uses per-lane online (m,s): max-update path
// is branched over (exec-masked) so steady state costs 1 exp per element.
// ---------------------------------------------------------------------------
__global__ __launch_bounds__(256) void score_lse_kernel(
    const _Float16* __restrict__ Qh, const _Float16* __restrict__ Kh,
    float* __restrict__ pm, float* __restrict__ ps)
{
    __shared__ _Float16 ktile[16 * LDSW];   // 16 K-rows x 256 (+8 pad)

    const int tid  = threadIdx.x;
    const int lane = tid & 31;
    const int wv   = tid >> 5;
    const int lg   = lane >> 4;
    const int ln   = lane & 15;
    const int srow = tid >> 4;              // staging row   (0..15)
    const int sch  = (tid & 15) * 16;       // staging chunk (halfs)

    const int split = blockIdx.x >> 6;                 // 0..7
    const int qrow  = (blockIdx.x & 63) * 128 + wv * 16;

    // Preload this wave's 16 Q rows as 8 A fragments (loop-resident)
    v16h a[8];
#pragma unroll
    for (int f = 0; f < 8; ++f) {
        const _Float16* p = Qh + (size_t)(qrow + ln) * DIM + f * 32 + lg * 16;
        v8h lo = *(const v8h*)(p);
        v8h hi = *(const v8h*)(p + 8);
        v16h t;
#pragma unroll
        for (int e = 0; e < 8; ++e) { t[e] = lo[e]; t[8 + e] = hi[e]; }
        a[f] = t;
    }

    float m[8], s[8];   // per-lane online LSE over this lane's columns
#pragma unroll
    for (int r = 0; r < 8; ++r) { m[r] = -__builtin_inff(); s[r] = 0.f; }

    const int nt0 = split * TILES_PER_SPLIT;

    // Pipeline prologue: fetch first K tile into registers
    v8h plo, phi;
    {
        const _Float16* src = Kh + (size_t)(nt0 * 16 + srow) * DIM + sch;
        plo = *(const v8h*)(src);
        phi = *(const v8h*)(src + 8);
    }

    for (int nt = nt0; nt < nt0 + TILES_PER_SPLIT; ++nt) {
        __syncthreads();   // previous tile fully consumed -> LDS reusable
        {   // publish pending tile to LDS
            v8h* dst = (v8h*)&ktile[srow * LDSW + sch];
            dst[0] = plo;
            dst[1] = phi;
        }
        __syncthreads();   // tile visible to all waves

        // Prefetch next tile: loads stay in flight across the WMMAs below
        if (nt + 1 < nt0 + TILES_PER_SPLIT) {
            const _Float16* src = Kh + (size_t)((nt + 1) * 16 + srow) * DIM + sch;
            plo = *(const v8h*)(src);
            phi = *(const v8h*)(src + 8);
        }

        v8f c = {};
#pragma unroll
        for (int f = 0; f < 8; ++f) {
            // B fragment: lane ln = col, k = f*32 + lg*16 + e (conflict-free)
            const _Float16* bp = &ktile[ln * LDSW + f * 32 + lg * 16];
            v8h blo = *(const v8h*)bp;
            v8h bhi = *(const v8h*)(bp + 8);
            v16h b;
#pragma unroll
            for (int e = 0; e < 8; ++e) { b[e] = blo[e]; b[8 + e] = bhi[e]; }
            c = __builtin_amdgcn_wmma_f32_16x16x32_f16(
                    false, a[f], false, b, (short)0, c, false, false);
        }

        // Per-lane online LSE: common path = 1 exp; rescale path branched over
#pragma unroll
        for (int r = 0; r < 8; ++r) {
            float v = c[r];
            if (v <= m[r]) {
                s[r] += __expf(v - m[r]);
            } else {
                s[r] = s[r] * __expf(m[r] - v) + 1.f;
                m[r] = v;
            }
        }
    }

    // Cross-lane merge within each 16-lane group (once per split)
#pragma unroll
    for (int r = 0; r < 8; ++r) {
        float mm = m[r];
        mm = fmaxf(mm, __shfl_xor(mm, 1, 32));
        mm = fmaxf(mm, __shfl_xor(mm, 2, 32));
        mm = fmaxf(mm, __shfl_xor(mm, 4, 32));
        mm = fmaxf(mm, __shfl_xor(mm, 8, 32));
        float sr = s[r] * __expf(m[r] - mm);
        sr += __shfl_xor(sr, 1, 32);
        sr += __shfl_xor(sr, 2, 32);
        sr += __shfl_xor(sr, 4, 32);
        sr += __shfl_xor(sr, 8, 32);
        if (ln == 0) {
            int row = qrow + r + lg * 8;
            pm[split * NQ + row] = mm;
            ps[split * NQ + row] = sr;
        }
    }
}

// ---------------------------------------------------------------------------
// Kernel 3: merge the 8 partial (m,s) per row -> lse, sum rows, negate.
// ---------------------------------------------------------------------------
__global__ __launch_bounds__(256) void finalize_kernel(
    const float* __restrict__ pm, const float* __restrict__ ps,
    float* __restrict__ out)
{
    __shared__ float sm[256];
    float acc = 0.f;
    for (int row = threadIdx.x; row < NQ; row += 256) {
        float mmax = -__builtin_inff();
#pragma unroll
        for (int sp = 0; sp < NSPLIT; ++sp)
            mmax = fmaxf(mmax, pm[sp * NQ + row]);
        float ssum = 0.f;
#pragma unroll
        for (int sp = 0; sp < NSPLIT; ++sp)
            ssum += ps[sp * NQ + row] * __expf(pm[sp * NQ + row] - mmax);
        acc += mmax + __logf(ssum);
    }
    sm[threadIdx.x] = acc;
    __syncthreads();
    for (int off = 128; off > 0; off >>= 1) {
        if (threadIdx.x < off) sm[threadIdx.x] += sm[threadIdx.x + off];
        __syncthreads();
    }
    if (threadIdx.x == 0) out[0] = -sm[0];
}

// ---------------------------------------------------------------------------
extern "C" void kernel_launch(void* const* d_in, const int* in_sizes, int n_in,
                              void* d_out, int out_size, void* d_ws, size_t ws_size,
                              hipStream_t stream)
{
    const float* xq = (const float*)d_in[0];
    const float* xk = (const float*)d_in[1];
    const float* Wq = (const float*)d_in[2];
    const float* Wk = (const float*)d_in[3];

    char* ws = (char*)d_ws;
    _Float16* Qh  = (_Float16*)ws;                                   // 4 MB
    _Float16* Kh  = Qh + (size_t)NQ * DIM;                           // 4 MB
    float*    pm  = (float*)(ws + 2u * NQ * DIM * sizeof(_Float16)); // 256 KB
    float*    ps  = pm + (size_t)NSPLIT * NQ;                        // 256 KB
    _Float16* Wqt = (_Float16*)(ps + (size_t)NSPLIT * NQ);           // 128 KB
    _Float16* Wkt = Wqt + (size_t)DIM * DIM;                         // 128 KB

    wconv_kernel<<<512, 256, 0, stream>>>(Wq, Wk, Wqt, Wkt);
    // 1024 row-tiles x 16 col-tiles over concat(Q,K) -> 2048 blocks of 8 waves
    proj_kernel<<<2048, 256, 0, stream>>>(xq, xk, Wqt, Wkt, Qh, Kh);
    // 64 row-blocks x 8 K-splits
    score_lse_kernel<<<64 * NSPLIT, 256, 0, stream>>>(Qh, Kh, pm, ps);
    finalize_kernel<<<1, 256, 0, stream>>>(pm, ps, (float*)d_out);
}
